// BlockwiseParallelTransformer_39195871543936
// MI455X (gfx1250) — matrix operand order repaired and checked by reading, not compiled, not run on hardware
//
#include <hip/hip_runtime.h>
#include <cstddef>
#include <cstdint>

// ---------------------------------------------------------------------------
// Types for CDNA5 WMMA (wave32): v_wmma_f32_16x16x32_bf16
// ---------------------------------------------------------------------------
typedef __attribute__((ext_vector_type(16))) __bf16 bf16x16;
typedef __attribute__((ext_vector_type(8)))  float  f32x8;
typedef __attribute__((__vector_size__(16))) int    v4i;

#define TB  2
#define TN  4096
#define TD  1024
#define THH 16
#define THD 64
#define TFF 4096
#define TM  (TB * TN)   // 8192 rows for all GEMMs

// ---------------------------------------------------------------------------
// gfx1250 async global->LDS copy (GLOBAL_LOAD_ASYNC_TO_LDS_B128, ASYNCcnt),
// guarded so the file compiles on any toolchain (fallback: sync copy).
// Builtin parameter types per hipcc diagnostic: (AS1 v4i*, AS3 v4i*, imm, imm)
// ---------------------------------------------------------------------------
#if defined(__gfx1250__) && __has_builtin(__builtin_amdgcn_global_load_async_to_lds_b128)
#define USE_ASYNC_LDS 1
#else
#define USE_ASYNC_LDS 0
#endif

__device__ __forceinline__ void copy_b128_to_lds(const __bf16* g, __bf16* l) {
#if USE_ASYNC_LDS
    __builtin_amdgcn_global_load_async_to_lds_b128(
        (__attribute__((address_space(1))) v4i*)g,
        (__attribute__((address_space(3))) v4i*)l, 0, 0);
#else
    *(uint4*)(void*)l = *(const uint4*)(const void*)g;
#endif
}

__device__ __forceinline__ void async_wait0() {
#if USE_ASYNC_LDS
#if __has_builtin(__builtin_amdgcn_s_wait_asynccnt)
    __builtin_amdgcn_s_wait_asynccnt(0);
#else
    asm volatile("s_wait_asynccnt 0x0" ::: "memory");
#endif
#endif
}

__device__ __forceinline__ f32x8 wmma_bf16f32(bf16x16 a, bf16x16 b, f32x8 c) {
    // D = A(16x32 bf16) * B(32x16 bf16) + C(16x16 f32)
    return __builtin_amdgcn_wmma_f32_16x16x32_bf16(false, a, false, b,
                                                   (short)0, c, false, false);
}

// A-matrix fragment (16x32 bf16), contiguous-k layout used here:
//  lane L holds row M = L%16; VGPR v holds K pair:
//   K = 2v + 16*(L/16)  (contiguous, matches load_b_frag)
__device__ __forceinline__ bf16x16 load_a_frag(const __bf16* base, int stride, int lane) {
    const int r  = lane & 15;
    const int hl = lane >> 4;
    union { bf16x16 v; unsigned int u[8]; } f;
    const __bf16* p = base + r * stride;
#pragma unroll
    for (int i = 0; i < 8; ++i) {
        const int k = 2 * i + 16 * hl;
        f.u[i] = *(const unsigned int*)(const void*)(p + k);
    }
    return f.v;
}

// B-matrix fragment (32x16 bf16), contiguous-k layout used here:
//  lane L holds column n = L%16; K = 16*(L/16) + e (e = 0..15 contiguous)
// base points at [nbase][kbase] of an LDS tile stored n-major (B transposed).
__device__ __forceinline__ bf16x16 load_b_frag(const __bf16* base, int stride, int lane) {
    const int n  = lane & 15;
    const int hl = lane >> 4;
    union { bf16x16 v; uint4 q[2]; } f;
    const uint4* p = (const uint4*)(const void*)(base + n * stride + 16 * hl);
    f.q[0] = p[0];
    f.q[1] = p[1];
    return f.v;
}

// ---------------------------------------------------------------------------
// Elementwise helpers
// ---------------------------------------------------------------------------
__global__ void cvt_bf16_kernel(const float* __restrict__ s, __bf16* __restrict__ d, long n) {
    long i      = (long)blockIdx.x * blockDim.x + threadIdx.x;
    long stride = (long)gridDim.x * blockDim.x;
    for (; i < n; i += stride) d[i] = (__bf16)s[i];
}

__global__ void addcvt_bf16_kernel(const float* __restrict__ a, const float* __restrict__ b,
                                   __bf16* __restrict__ d, long n) {
    long i      = (long)blockIdx.x * blockDim.x + threadIdx.x;
    long stride = (long)gridDim.x * blockDim.x;
    for (; i < n; i += stride) d[i] = (__bf16)(a[i] + b[i]);
}

// ---------------------------------------------------------------------------
// WMMA GEMM:  out = epilogue( scale * (A[MxK] @ B[KxN] + bias[N]) )
//   mode 0: outB = bf16(val)             (Q/K/V projections)
//   mode 1: outB = bf16(max(val,0))      (FFN layer 1, ReLU)
//   mode 2: outF = val + res1 + res2     (FFN layer 2 + residuals)
// Block tile 128x128, BK=32, 256 threads = 8 waves, wave tile 32x64.
// Double-buffered LDS: one barrier per K-step; A tile staged with async
// global->LDS (overlaps the WMMA stream), B tile transposed manually.
// ---------------------------------------------------------------------------
__global__ __launch_bounds__(256) void gemm_kernel(
    const __bf16* __restrict__ A, const __bf16* __restrict__ Bm,
    const float* __restrict__ bias, float scale,
    int M, int N, int K, int mode,
    const float* __restrict__ res1, const float* __restrict__ res2,
    __bf16* __restrict__ outB, float* __restrict__ outF)
{
    __shared__ __align__(16) __bf16 Al[2][128 * 40];   // [buf][row][k], stride 40
    __shared__ __align__(16) __bf16 Bl[2][128 * 40];   // [buf][n][k]  (B transposed)

    const int tid  = threadIdx.x;
    const int lane = tid & 31;
    const int w    = tid >> 5;
    const int wr   = w >> 1;          // 0..3 -> 32-row slice
    const int wc   = w & 1;           // 0..1 -> 64-col slice
    const int m0   = blockIdx.y * 128;
    const int n0   = blockIdx.x * 128;

    // staging geometry
    const int arow = tid >> 1, ac0 = (tid & 1) * 16;   // A: 128 rows x 32 k
    const int bk   = tid >> 3, bc0 = (tid & 7) * 16;   // B: 32 k x 128 n

    f32x8 acc[2][4] = {};

    // ---- prologue: stage buffer 0 with k0 = 0 ----
    {
        const __bf16* ga = A + (size_t)(m0 + arow) * K + ac0;
        copy_b128_to_lds(ga,     &Al[0][arow * 40 + ac0]);
        copy_b128_to_lds(ga + 8, &Al[0][arow * 40 + ac0 + 8]);
        union { uint4 q[2]; __bf16 e[16]; } t;
        const uint4* s = (const uint4*)(const void*)(Bm + (size_t)bk * N + n0 + bc0);
        t.q[0] = s[0]; t.q[1] = s[1];
#pragma unroll
        for (int i = 0; i < 16; ++i) Bl[0][(bc0 + i) * 40 + bk] = t.e[i];
    }
    async_wait0();
    __syncthreads();

    int buf = 0;
    for (int k0 = 0; k0 < K; k0 += 32) {
        // ---- stage next tile into buf^1 while computing on buf ----
        const int kn = k0 + 32;
        if (kn < K) {
            const __bf16* ga = A + (size_t)(m0 + arow) * K + kn + ac0;
            copy_b128_to_lds(ga,     &Al[buf ^ 1][arow * 40 + ac0]);
            copy_b128_to_lds(ga + 8, &Al[buf ^ 1][arow * 40 + ac0 + 8]);
            union { uint4 q[2]; __bf16 e[16]; } t;
            const uint4* s = (const uint4*)(const void*)(Bm + (size_t)(kn + bk) * N + n0 + bc0);
            t.q[0] = s[0]; t.q[1] = s[1];
#pragma unroll
            for (int i = 0; i < 16; ++i) Bl[buf ^ 1][(bc0 + i) * 40 + bk] = t.e[i];
        }

        // ---- compute on current buffer ----
        bf16x16 af[2], bfr[4];
#pragma unroll
        for (int rt = 0; rt < 2; ++rt)
            af[rt] = load_a_frag(&Al[buf][(wr * 32 + rt * 16) * 40], 40, lane);
#pragma unroll
        for (int ct = 0; ct < 4; ++ct)
            bfr[ct] = load_b_frag(&Bl[buf][(wc * 64 + ct * 16) * 40], 40, lane);
#pragma unroll
        for (int rt = 0; rt < 2; ++rt)
#pragma unroll
            for (int ct = 0; ct < 4; ++ct)
                acc[rt][ct] = wmma_bf16f32(af[rt], bfr[ct], acc[rt][ct]);

        async_wait0();
        __syncthreads();
        buf ^= 1;
    }

    // Epilogue. C layout: lane holds col n0.. + l16, rows v + 8*(lane/16).
    const int hl = lane >> 4, l16 = lane & 15;
#pragma unroll
    for (int rt = 0; rt < 2; ++rt)
#pragma unroll
        for (int ct = 0; ct < 4; ++ct) {
            const int col  = n0 + wc * 64 + ct * 16 + l16;
            const float bb = bias[col];
#pragma unroll
            for (int v = 0; v < 8; ++v) {
                const int row    = m0 + wr * 32 + rt * 16 + v + 8 * hl;
                const size_t idx = (size_t)row * N + col;
                const float val  = (acc[rt][ct][v] + bb) * scale;
                if (mode == 0)      outB[idx] = (__bf16)val;
                else if (mode == 1) outB[idx] = (__bf16)fmaxf(val, 0.0f);
                else                outF[idx] = val + res1[idx] + res2[idx];
            }
        }
}

// ---------------------------------------------------------------------------
// Blockwise attention. One WG = (b, head, 64 q-rows); 4 waves x 16 q-rows.
// Per KV block (512 keys): flash-style running max over 64-key chunks
// (== reference per-block max). Totals folded per block, NO cross-block
// rescale (matches reference semantics exactly).
// ---------------------------------------------------------------------------
__global__ __launch_bounds__(128) void attn_kernel(
    const __bf16* __restrict__ Qg, const __bf16* __restrict__ Kg,
    const __bf16* __restrict__ Vg, float* __restrict__ attnO)
{
    __shared__ __align__(16) __bf16 Qs [64 * 72];        // [qrow][hd]
    __shared__ __align__(16) __bf16 Ks [64 * 72];        // [key][hd]   (B-layout for QK^T)
    __shared__ __align__(16) __bf16 Vts[64 * 72];        // [hd][key]   (B-layout for P*V)
    __shared__ __align__(16) __bf16 Pb [4 * 16 * 72];    // per-wave P scratch [qrow][key]

    const int tid  = threadIdx.x;
    const int lane = tid & 31;
    const int w    = tid >> 5;
    const int hl   = lane >> 4;
    const int l16  = lane & 15;
    const int qbase = blockIdx.x * 64;
    const int h     = blockIdx.y;
    const int b     = blockIdx.z;
    const size_t baseBH = (size_t)b * TN * TD + (size_t)h * THD;

    const int srow = tid >> 1, sd0 = (tid & 1) * 32;     // staging geometry

    {   // stage the 64x64 Q tile once (async; completion folded into first chunk wait)
        const __bf16* gq = Qg + baseBH + (size_t)(qbase + srow) * TD + sd0;
        __bf16* lq = &Qs[srow * 72 + sd0];
#pragma unroll
        for (int i = 0; i < 4; ++i) copy_b128_to_lds(gq + 8 * i, lq + 8 * i);
    }

    f32x8 numT[4] = {}, nblk[4] = {};
    float denT[8], denB[8], mrow[8];
#pragma unroll
    for (int v = 0; v < 8; ++v) { denT[v] = 0.0f; denB[v] = 0.0f; mrow[v] = -__builtin_inff(); }

    for (int kb = 0; kb < 8; ++kb) {
        for (int ch = 0; ch < 8; ++ch) {
            const int keyb = kb * 512 + ch * 64;
            __syncthreads();    // protect K/V tiles from previous chunk's readers
            {   // stage K chunk [64 keys][64 hd] (async) and V transposed [64 hd][64 keys]
                const __bf16* gk = Kg + baseBH + (size_t)(keyb + srow) * TD + sd0;
                __bf16* lk = &Ks[srow * 72 + sd0];
#pragma unroll
                for (int i = 0; i < 4; ++i) copy_b128_to_lds(gk + 8 * i, lk + 8 * i);
                union { uint4 q[4]; __bf16 e[32]; } tv;
                const uint4* sv = (const uint4*)(const void*)(Vg + baseBH + (size_t)(keyb + srow) * TD + sd0);
                tv.q[0] = sv[0]; tv.q[1] = sv[1]; tv.q[2] = sv[2]; tv.q[3] = sv[3];
#pragma unroll
                for (int i = 0; i < 32; ++i) Vts[(sd0 + i) * 72 + srow] = tv.e[i];
            }
            async_wait0();
            __syncthreads();

            // ---- S = Q * K^T  (16 q-rows x 64 keys per wave) ----
            bf16x16 aq0 = load_a_frag(&Qs[(w * 16) * 72 + 0],  72, lane);
            bf16x16 aq1 = load_a_frag(&Qs[(w * 16) * 72 + 32], 72, lane);
            f32x8 sacc[4] = {};
#pragma unroll
            for (int kt = 0; kt < 4; ++kt) {
                bf16x16 b0 = load_b_frag(&Ks[(kt * 16) * 72 + 0],  72, lane);
                sacc[kt] = wmma_bf16f32(aq0, b0, sacc[kt]);
                bf16x16 b1 = load_b_frag(&Ks[(kt * 16) * 72 + 32], 72, lane);
                sacc[kt] = wmma_bf16f32(aq1, b1, sacc[kt]);
            }

            // ---- per-row chunk max (reduce 4 tiles, then 16 lanes of the half) ----
            float mx[8];
#pragma unroll
            for (int v = 0; v < 8; ++v)
                mx[v] = fmaxf(fmaxf(sacc[0][v], sacc[1][v]), fmaxf(sacc[2][v], sacc[3][v]));
#pragma unroll
            for (int mk = 1; mk < 16; mk <<= 1)
#pragma unroll
                for (int v = 0; v < 8; ++v)
                    mx[v] = fmaxf(mx[v], __shfl_xor(mx[v], mk, 32));

            float mnew[8], corr[8], rs[8];
#pragma unroll
            for (int v = 0; v < 8; ++v) {
                mnew[v] = fmaxf(mrow[v], mx[v]);
                corr[v] = __expf(mrow[v] - mnew[v]);   // first chunk: exp(-inf) = 0
                mrow[v] = mnew[v];
                rs[v]   = 0.0f;
            }

            // ---- P = exp(S - m); stash bf16 P in A-layout-friendly LDS ----
#pragma unroll
            for (int kt = 0; kt < 4; ++kt)
#pragma unroll
                for (int v = 0; v < 8; ++v) {
                    const float p = __expf(sacc[kt][v] - mnew[v]);
                    rs[v] += p;
                    Pb[(w * 16 + v + 8 * hl) * 72 + kt * 16 + l16] = (__bf16)p;
                }
#pragma unroll
            for (int mk = 1; mk < 16; mk <<= 1)
#pragma unroll
                for (int v = 0; v < 8; ++v)
                    rs[v] += __shfl_xor(rs[v], mk, 32);
#pragma unroll
            for (int v = 0; v < 8; ++v) denB[v] = denB[v] * corr[v] + rs[v];
#pragma unroll
            for (int nt = 0; nt < 4; ++nt)
#pragma unroll
                for (int v = 0; v < 8; ++v) nblk[nt][v] *= corr[v];

            // ---- num += P * V ----
            bf16x16 ap0 = load_a_frag(&Pb[(w * 16) * 72 + 0],  72, lane);
            bf16x16 ap1 = load_a_frag(&Pb[(w * 16) * 72 + 32], 72, lane);
#pragma unroll
            for (int nt = 0; nt < 4; ++nt) {
                bf16x16 bv0 = load_b_frag(&Vts[(nt * 16) * 72 + 0],  72, lane);
                nblk[nt] = wmma_bf16f32(ap0, bv0, nblk[nt]);
                bf16x16 bv1 = load_b_frag(&Vts[(nt * 16) * 72 + 32], 72, lane);
                nblk[nt] = wmma_bf16f32(ap1, bv1, nblk[nt]);
            }
        }
        // ---- fold this KV block into totals; reset block state (no rescale) ----
#pragma unroll
        for (int nt = 0; nt < 4; ++nt) {
            numT[nt] += nblk[nt];
#pragma unroll
            for (int v = 0; v < 8; ++v) nblk[nt][v] = 0.0f;
        }
#pragma unroll
        for (int v = 0; v < 8; ++v) {
            denT[v] += denB[v];
            denB[v]  = 0.0f;
            mrow[v]  = -__builtin_inff();
        }
    }

    // attn = num / den  -> f32 buffer laid out (b, N, H*HD)
#pragma unroll
    for (int nt = 0; nt < 4; ++nt)
#pragma unroll
        for (int v = 0; v < 8; ++v) {
            const int row    = qbase + w * 16 + v + 8 * hl;
            const size_t idx = (size_t)b * TN * TD + (size_t)row * TD
                             + (size_t)h * THD + nt * 16 + l16;
            attnO[idx] = numT[nt][v] / denT[v];
        }
}

// ---------------------------------------------------------------------------
// Launch
// ---------------------------------------------------------------------------
extern "C" void kernel_launch(void* const* d_in, const int* in_sizes, int n_in,
                              void* d_out, int out_size, void* d_ws, size_t ws_size,
                              hipStream_t stream) {
    (void)in_sizes; (void)n_in; (void)out_size; (void)ws_size;
    const float* x  = (const float*)d_in[0];
    const float* Wq = (const float*)d_in[1];
    const float* bq = (const float*)d_in[2];
    const float* Wk = (const float*)d_in[3];
    const float* bk = (const float*)d_in[4];
    const float* Wv = (const float*)d_in[5];
    const float* bv = (const float*)d_in[6];
    const float* W1 = (const float*)d_in[7];
    const float* b1 = (const float*)d_in[8];
    const float* W2 = (const float*)d_in[9];
    const float* b2 = (const float*)d_in[10];
    float* out = (float*)d_out;

    char* ws = (char*)d_ws;
    size_t off = 0;
    auto take = [&](size_t bytes) -> char* {
        char* p = ws + off;
        off += (bytes + 255) & ~(size_t)255;
        return p;
    };
    __bf16* xb    = (__bf16*)take((size_t)TM * TD * 2);
    __bf16* wqb   = (__bf16*)take((size_t)TD * TD * 2);
    __bf16* wkb   = (__bf16*)take((size_t)TD * TD * 2);
    __bf16* wvb   = (__bf16*)take((size_t)TD * TD * 2);
    __bf16* w1b   = (__bf16*)take((size_t)TD * TFF * 2);
    __bf16* w2b   = (__bf16*)take((size_t)TFF * TD * 2);
    __bf16* Qb    = (__bf16*)take((size_t)TM * TD * 2);
    __bf16* Kb    = (__bf16*)take((size_t)TM * TD * 2);
    __bf16* Vb    = (__bf16*)take((size_t)TM * TD * 2);
    float*  attnb = (float*) take((size_t)TM * TD * 4);
    __bf16* f1b   = (__bf16*)take((size_t)TM * TFF * 2);
    __bf16* hb    = xb;   // reuse: x_bf16 is dead after the QKV projections

    // bf16 conversions
    cvt_bf16_kernel<<<2048, 256, 0, stream>>>(x,  xb,  (long)TM * TD);
    cvt_bf16_kernel<<<512,  256, 0, stream>>>(Wq, wqb, (long)TD * TD);
    cvt_bf16_kernel<<<512,  256, 0, stream>>>(Wk, wkb, (long)TD * TD);
    cvt_bf16_kernel<<<512,  256, 0, stream>>>(Wv, wvb, (long)TD * TD);
    cvt_bf16_kernel<<<1024, 256, 0, stream>>>(W1, w1b, (long)TD * TFF);
    cvt_bf16_kernel<<<1024, 256, 0, stream>>>(W2, w2b, (long)TFF * TD);

    // Q/K/V projections (Q pre-scaled by 1/sqrt(HD) = 0.125)
    dim3 gproj(TD / 128, TM / 128);
    gemm_kernel<<<gproj, 256, 0, stream>>>(xb, wqb, bq, 0.125f, TM, TD, TD, 0,
                                           nullptr, nullptr, Qb, nullptr);
    gemm_kernel<<<gproj, 256, 0, stream>>>(xb, wkb, bk, 1.0f,   TM, TD, TD, 0,
                                           nullptr, nullptr, Kb, nullptr);
    gemm_kernel<<<gproj, 256, 0, stream>>>(xb, wvb, bv, 1.0f,   TM, TD, TD, 0,
                                           nullptr, nullptr, Vb, nullptr);

    // blockwise attention
    dim3 ag(TN / 64, THH, TB);
    attn_kernel<<<ag, 128, 0, stream>>>(Qb, Kb, Vb, attnb);

    // h = bf16(attn + x)   (reuses xb storage)
    addcvt_bf16_kernel<<<2048, 256, 0, stream>>>(attnb, x, hb, (long)TM * TD);

    // FFN1: relu(h @ W1 + b1) -> bf16
    dim3 g1(TFF / 128, TM / 128);
    gemm_kernel<<<g1, 256, 0, stream>>>(hb, w1b, b1, 1.0f, TM, TFF, TD, 1,
                                        nullptr, nullptr, f1b, nullptr);
    // FFN2 fused epilogue: out = (f1 @ W2 + b2) + attn + x
    dim3 g2(TD / 128, TM / 128);
    gemm_kernel<<<g2, 256, 0, stream>>>(f1b, w2b, b2, 1.0f, TM, TD, TFF, 2,
                                        attnb, x, nullptr, out);
}
